// EdgeConvBlock_9715216023597
// MI455X (gfx1250) — compile-verified, matrix operand order checked by reading, not run on hardware
//
#include <hip/hip_runtime.h>
#include <hip/hip_bf16.h>

typedef __attribute__((ext_vector_type(16))) _Float16 v16h;
typedef __attribute__((ext_vector_type(8)))  _Float16 v8h;
typedef __attribute__((ext_vector_type(4)))  _Float16 v4h;
typedef __attribute__((ext_vector_type(8)))  float    v8f;

#define LDSTRIDE 136   // 128 + 8 halves pad: row step = 68 dwords -> 4-bank rotation, conflict-free b128 reads
#define OSTRIDE  132   // fp32 out-staging stride: 128 + 4 floats pad
#define BN_EPS 1e-5f

// ---------------------------------------------------------------------------
// Kernel 0: zero the aggregation output region and the ws accumulators
// ---------------------------------------------------------------------------
__global__ void zero_kernel(float* __restrict__ a, long long na,
                            float* __restrict__ b, long long nb) {
    long long i = (long long)blockIdx.x * blockDim.x + threadIdx.x;
    if (i < na) a[i] = 0.0f;
    if (i < nb) b[i] = 0.0f;
}

// ---------------------------------------------------------------------------
// Kernel 1: fused gather + GEMM via WMMA.
//   h[e][c] = sum_k concat(x[dst[e]], x[src[e]])[k] * W1[c][k] + b1[c]
// Block: 256 threads (8 waves), tile = 128 edges x 128 channels, K=256.
// K processed in two 128-halves (half0 = x[dst], half1 = x[src]) staged f16
// in LDS. Per k-step all 8 B-fragments are preloaded into registers so the
// 8 WMMAs issue back-to-back behind a single dscnt wait. Output is staged
// through LDS (fp32) for fully coalesced b128 stores of raw h (+bias).
// ---------------------------------------------------------------------------
__global__ __launch_bounds__(256) void edgeconv_gemm_kernel(
    const float* __restrict__ x,      // [N,128]
    const int*   __restrict__ eidx,   // [2,E] (row0 = src, row1 = dst)
    const float* __restrict__ W1,     // [128,256]
    const float* __restrict__ b1,     // [128]
    float*       __restrict__ hbuf,   // [E,128]
    long long E)
{
    // 2 f16 tiles (2*34816 B) during compute; reused as one fp32 staging
    // tile (128*132*4 = 67584 B) for the coalesced output phase.
    __shared__ __attribute__((aligned(16))) unsigned char smem[2 * 128 * LDSTRIDE * 2];
    _Float16* sA  = (_Float16*)smem;                           // [128][LDSTRIDE]
    _Float16* sW  = (_Float16*)(smem + 128 * LDSTRIDE * 2);    // [128][LDSTRIDE]
    float*    sOut = (float*)smem;                             // [128][OSTRIDE]

    const int  tid   = threadIdx.x;
    const int  wave  = tid >> 5;          // 0..7
    const int  lane  = tid & 31;          // wave32
    const int  kg    = lane >> 4;         // k-group (0/1) per A/B VGPR layout
    const int  ln    = lane & 15;         // row-in-tile / col-in-tile
    const int  rw    = wave * 16;         // this wave's 16 rows
    const long long eBase = (long long)blockIdx.x * 128;

    v8f acc[8];
    #pragma unroll
    for (int nt = 0; nt < 8; ++nt) acc[nt] = (v8f){0,0,0,0,0,0,0,0};

    #pragma unroll
    for (int ko = 0; ko < 2; ++ko) {
        const int k0 = ko * 128;

        // --- stage W half-tile: sW[n][k] = (f16) W1[n][k0+k], k in [0,128)
        {
            const int n     = tid >> 1;
            const int halfk = (tid & 1) * 64;
            const float4* wrow = (const float4*)(W1 + (size_t)n * 256 + k0 + halfk);
            _Float16* dp = &sW[n * LDSTRIDE + halfk];
            #pragma unroll
            for (int j = 0; j < 16; ++j) {
                float4 f = wrow[j];
                v4h h4 = { (_Float16)f.x, (_Float16)f.y, (_Float16)f.z, (_Float16)f.w };
                *(v4h*)(dp + j * 4) = h4;
            }
        }
        // --- stage A half-tile: gather x[dst] (ko==0) or x[src] (ko==1)
        {
            const int r     = tid >> 1;
            const int halfk = (tid & 1) * 64;
            long long e  = eBase + r;
            long long ec = e < E ? e : (E - 1);
            const int node = (ko == 0) ? eidx[E + ec] : eidx[ec];
            const float4* xrow = (const float4*)(x + (size_t)node * 128 + halfk);
            _Float16* dp = &sA[r * LDSTRIDE + halfk];
            #pragma unroll
            for (int j = 0; j < 16; ++j) {
                float4 f = xrow[j];
                v4h h4 = { (_Float16)f.x, (_Float16)f.y, (_Float16)f.z, (_Float16)f.w };
                *(v4h*)(dp + j * 4) = h4;
            }
        }
        __syncthreads();

        // --- 4 k-steps of 32 over this 128-half
        #pragma unroll
        for (int kt = 0; kt < 4; ++kt) {
            const int cb = kt * 32 + kg * 8;      // per-lane half-chunk base
            const int arow = rw + ln;
            v8h alo = *(const v8h*)&sA[arow * LDSTRIDE + cb];
            v8h ahi = *(const v8h*)&sA[arow * LDSTRIDE + cb + 16];
            v16h afrag = __builtin_shufflevector(alo, ahi,
                0,1,2,3,4,5,6,7,8,9,10,11,12,13,14,15);

            // Preload all 8 B fragments -> one dscnt wait, 8 WMMAs back-to-back
            v16h bfrag[8];
            #pragma unroll
            for (int nt = 0; nt < 8; ++nt) {
                const int bcol = nt * 16 + ln;
                v8h blo = *(const v8h*)&sW[bcol * LDSTRIDE + cb];
                v8h bhi = *(const v8h*)&sW[bcol * LDSTRIDE + cb + 16];
                bfrag[nt] = __builtin_shufflevector(blo, bhi,
                    0,1,2,3,4,5,6,7,8,9,10,11,12,13,14,15);
            }
            #pragma unroll
            for (int nt = 0; nt < 8; ++nt) {
                acc[nt] = __builtin_amdgcn_wmma_f32_16x16x32_f16(
                    false, afrag, false, bfrag[nt], (short)0, acc[nt], false, false);
            }
        }
        __syncthreads();
    }

    // --- bias + transpose through LDS for coalesced output.
    // C/D layout: lane L, VGPR v -> row rw+(L>>4)*8+v, col (L&15)
    #pragma unroll
    for (int nt = 0; nt < 8; ++nt) {
        const int col = nt * 16 + ln;
        const float bias = b1[col];
        #pragma unroll
        for (int v = 0; v < 8; ++v) {
            const int row = rw + kg * 8 + v;
            sOut[row * OSTRIDE + col] = acc[nt][v] + bias;
        }
    }
    __syncthreads();

    // Each wave stores rows wave, wave+8, ..., wave+120: one fully-coalesced
    // 512B global_store_b128 per wave per row.
    #pragma unroll
    for (int it = 0; it < 16; ++it) {
        const int row = it * 8 + wave;
        const long long e = eBase + row;
        if (e < E) {
            float4 vv = *(const float4*)&sOut[row * OSTRIDE + lane * 4];
            *(float4*)(hbuf + (size_t)e * 128 + lane * 4) = vv;
        }
    }
}

// ---------------------------------------------------------------------------
// Kernel 2: per-channel sum / sumsq over edges (for BatchNorm batch stats)
// ---------------------------------------------------------------------------
__global__ __launch_bounds__(128) void stats_kernel(
    const float* __restrict__ hbuf, float* __restrict__ sum,
    float* __restrict__ sumsq, long long E)
{
    const int c = threadIdx.x;   // 128 channels
    float s = 0.0f, s2 = 0.0f;
    for (long long e = blockIdx.x; e < E; e += gridDim.x) {
        float v = hbuf[(size_t)e * 128 + c];
        s  += v;
        s2 += v * v;
    }
    atomicAdd(&sum[c], s);
    atomicAdd(&sumsq[c], s2);
}

// ---------------------------------------------------------------------------
// Kernel 3: fold stats into per-channel scale/shift
// ---------------------------------------------------------------------------
__global__ void finalize_stats_kernel(
    const float* __restrict__ sum, const float* __restrict__ sumsq,
    const float* __restrict__ gamma, const float* __restrict__ beta,
    float* __restrict__ scale, float* __restrict__ shift, long long E)
{
    const int c = threadIdx.x;
    const float invE = 1.0f / (float)E;
    const float mean = sum[c] * invE;
    const float var  = fmaxf(sumsq[c] * invE - mean * mean, 0.0f);
    const float sc   = gamma[c] * rsqrtf(var + BN_EPS);
    scale[c] = sc;
    shift[c] = beta[c] - mean * sc;
}

// ---------------------------------------------------------------------------
// Kernel 4: normalize + ReLU in place, scatter-add to nodes + counts.
// One 32-thread group per edge; each thread handles a float4 of channels.
// ---------------------------------------------------------------------------
__global__ __launch_bounds__(256) void norm_scatter_kernel(
    float* __restrict__ hbuf,          // [E,128] in/out (h -> edge_act)
    const int* __restrict__ eidx,      // [2,E]
    const float* __restrict__ scale, const float* __restrict__ shift,
    float* __restrict__ agg,           // [N,128] (pre-zeroed)
    float* __restrict__ cnt,           // [N]     (pre-zeroed)
    long long E)
{
    const long long gid = (long long)blockIdx.x * blockDim.x + threadIdx.x;
    const long long e = gid >> 5;
    if (e >= E) return;
    const int q  = (int)(gid & 31);
    const int c0 = q * 4;
    const int dst = eidx[E + e];

    float4 h = *(const float4*)(hbuf + (size_t)e * 128 + c0);
    float4 sc = *(const float4*)(scale + c0);
    float4 sh = *(const float4*)(shift + c0);
    float4 a;
    a.x = fmaxf(h.x * sc.x + sh.x, 0.0f);
    a.y = fmaxf(h.y * sc.y + sh.y, 0.0f);
    a.z = fmaxf(h.z * sc.z + sh.z, 0.0f);
    a.w = fmaxf(h.w * sc.w + sh.w, 0.0f);
    *(float4*)(hbuf + (size_t)e * 128 + c0) = a;

    float* ap = agg + (size_t)dst * 128 + c0;
    atomicAdd(ap + 0, a.x);
    atomicAdd(ap + 1, a.y);
    atomicAdd(ap + 2, a.z);
    atomicAdd(ap + 3, a.w);
    if (q == 0) atomicAdd(&cnt[dst], 1.0f);
}

// ---------------------------------------------------------------------------
// Kernel 5: divide by counts (nodes with no in-edges -> 0 via max(cnt,1))
// ---------------------------------------------------------------------------
__global__ void mean_kernel(float* __restrict__ agg, const float* __restrict__ cnt,
                            long long N)
{
    const long long i = (long long)blockIdx.x * blockDim.x + threadIdx.x;
    if (i >= N * 128) return;
    const long long n = i >> 7;
    agg[i] = agg[i] / fmaxf(cnt[n], 1.0f);
}

// ---------------------------------------------------------------------------
extern "C" void kernel_launch(void* const* d_in, const int* in_sizes, int n_in,
                              void* d_out, int out_size, void* d_ws, size_t ws_size,
                              hipStream_t stream) {
    const float* x     = (const float*)d_in[0];
    const int*   eidx  = (const int*)  d_in[1];   // [2,E]: row0=src, row1=dst
    const float* W1    = (const float*)d_in[2];   // [128,256]
    const float* b1    = (const float*)d_in[3];
    const float* gamma = (const float*)d_in[4];
    const float* beta  = (const float*)d_in[5];

    const long long N = (long long)in_sizes[0] / 128;
    const long long E = (long long)in_sizes[1] / 2;

    float* out_agg = (float*)d_out;                        // [N,128]
    float* hbuf    = (float*)d_out + (size_t)N * 128;      // [E,128] edge_act region

    // ws layout: [0,N) cnt | [N,N+128) sum | [N+128,N+256) sumsq
    //            [N+256,N+384) scale | [N+384,N+512) shift
    float* ws    = (float*)d_ws;
    float* cnt   = ws;
    float* sum   = ws + N;
    float* sumsq = ws + N + 128;
    float* scale = ws + N + 256;
    float* shift = ws + N + 384;

    // 0) zero accumulation buffers
    {
        long long na = N * 128, nb = N + 256;
        long long mx = na > nb ? na : nb;
        int blocks = (int)((mx + 255) / 256);
        zero_kernel<<<blocks, 256, 0, stream>>>(out_agg, na, cnt, nb);
    }
    // 1) WMMA GEMM -> raw h (+bias) into hbuf
    {
        int blocks = (int)((E + 127) / 128);
        edgeconv_gemm_kernel<<<blocks, 256, 0, stream>>>(x, eidx, W1, b1, hbuf, E);
    }
    // 2) BN batch statistics
    stats_kernel<<<2048, 128, 0, stream>>>(hbuf, sum, sumsq, E);
    // 3) fold into scale/shift
    finalize_stats_kernel<<<1, 128, 0, stream>>>(sum, sumsq, gamma, beta, scale, shift, E);
    // 4) normalize + ReLU in place, scatter-add + counts
    {
        long long total = E * 32;
        int blocks = (int)((total + 255) / 256);
        norm_scatter_kernel<<<blocks, 256, 0, stream>>>(hbuf, eidx, scale, shift,
                                                        out_agg, cnt, E);
    }
    // 5) scatter-mean finalize
    {
        long long total = N * 128;
        int blocks = (int)((total + 255) / 256);
        mean_kernel<<<blocks, 256, 0, stream>>>(out_agg, cnt, N);
    }
}
